// SimpleInstanceSeg_20212116095211
// MI455X (gfx1250) — compile-verified, matrix operand order checked by reading, not run on hardware
//
#include <hip/hip_runtime.h>
#include <hip/hip_bf16.h>
#include <cfloat>

typedef __attribute__((ext_vector_type(16))) _Float16 v16h;
typedef __attribute__((ext_vector_type(8)))  _Float16 h8;
typedef __attribute__((ext_vector_type(8)))  float    v8f;

#define BATCH   8
#define IMG_H   256
#define IMG_W   512
#define FH      32
#define FW      64
#define NA      2048      // anchors per image = FH*FW
#define NTOT    (BATCH*NA)
#define NCLS    21        // NUM_CLASSES+1
#define PCOEF   32
#define TOPK    100
#define TOPKP   112       // TOPK padded to 16 for WMMA
#define IOU_THR 0.5f
#define ZPAGE_H 512       // zero page: 512 f16 (covers Cin<=256 + frag offsets)

// ---------------------------------------------------------------------------
__global__ void zero_init_kernel(_Float16* __restrict__ z, int n)
{
    const int i = blockIdx.x * blockDim.x + threadIdx.x;
    if (i < n) z[i] = (_Float16)0.0f;
}

// ---------------------------------------------------------------------------
// Weight repack: w[Cout][Cin][3][3] (f32) -> wr[CoutPad][taps*Cin] (f16) with
// K ordered (kh,kw,cin) and zero-padded rows, so A-fragments need no guards.
// ---------------------------------------------------------------------------
__global__ void repack_w_kernel(const float* __restrict__ w,
                                _Float16* __restrict__ wr,
                                int Cout, int CoutPad, int Cin, int taps)
{
    const int K = taps * Cin;
    const long long total = (long long)CoutPad * K;
    const long long idx = (long long)blockIdx.x * blockDim.x + threadIdx.x;
    if (idx >= total) return;
    const int m = (int)(idx / K);
    const int k = (int)(idx - (long long)m * K);
    float v = 0.0f;
    if (m < Cout) {
        const int t = k / Cin;           // kh*3+kw
        const int c = k - t * Cin;
        v = w[((long long)m * Cin + c) * taps + t];
    }
    wr[idx] = (_Float16)v;
}

// ---------------------------------------------------------------------------
// conv1: 3->64 s2 direct VALU conv (K=27, bandwidth-trivial), emits NHWC f16.
// ---------------------------------------------------------------------------
__global__ void conv1_direct_kernel(const float* __restrict__ img,  // [B][3][256][512]
                                    const float* __restrict__ w1,   // [64][3][3][3]
                                    const float* __restrict__ b1,
                                    _Float16* __restrict__ out)     // [B][128][256][64]
{
    const long long idx = (long long)blockIdx.x * blockDim.x + threadIdx.x;
    const long long total = 8LL * 128 * 256 * 64;
    if (idx >= total) return;
    const int co = (int)(idx & 63);
    long long t = idx >> 6;
    const int ow = (int)(t & 255); t >>= 8;
    const int oh = (int)(t & 127);
    const int b  = (int)(t >> 7);

    float acc = b1[co];
#pragma unroll
    for (int kh = 0; kh < 3; ++kh) {
        const int ih = oh * 2 + kh - 1;
        if (ih < 0 || ih >= 256) continue;
#pragma unroll
        for (int kw2 = 0; kw2 < 3; ++kw2) {
            const int iw = ow * 2 + kw2 - 1;
            if (iw < 0 || iw >= 512) continue;
#pragma unroll
            for (int ci = 0; ci < 3; ++ci)
                acc += img[((long long)(b * 3 + ci) * 256 + ih) * 512 + iw]
                     * w1[((co * 3 + ci) * 3 + kh) * 3 + kw2];
        }
    }
    out[idx] = (_Float16)fmaxf(acc, 0.0f);
}

// ---------------------------------------------------------------------------
// Fast implicit-GEMM conv, NHWC f16 activations, repacked f16 weights.
// One wave = 16(M) x 64(N) tile: 4 accumulators share one A fragment per
// K-chunk (4x arithmetic intensity on A). K-chunks of 32 = one tap slice of
// 32 input channels -> every fragment is two aligned b128 loads. Boundary
// columns read a zero page instead of branching (no exec masking, no
// zero-fill movs in the steady-state loop).
// A-frag (16x32): lane row m=lane&15; elements [0..7]=K off half*8,
// [8..15]=K off 16+half*8 (ISA 7.12.2). B-frag mirrors with lane col n.
// C/D: row = r + 8*(lane>>4), col = lane&15.  Requires Cin % 32 == 0.
// ---------------------------------------------------------------------------
__global__ __launch_bounds__(32) void wmma_conv_nhwc_kernel(
    const _Float16* __restrict__ x,   // [B][Hin][Win][Cin]
    const _Float16* __restrict__ wr,  // [CoutPad][taps*Cin]
    const float* __restrict__ bias,   // [Cout]
    const _Float16* __restrict__ zpage,
    void* __restrict__ out, int out_f32,  // NHWC: [N][CoutPad] f16 or f32
    int Cin, int Cout, int CoutPad, int Hin, int Win, int Ho, int Wo,
    int stride, int pad, int taps, int relu)
{
    const int lane = threadIdx.x;
    const int half = lane >> 4;
    const int l15  = lane & 15;
    const int mtile = blockIdx.x;
    const int ntile = blockIdx.y;     // 64-wide column tile
    const int HoWo  = Ho * Wo;
    const int K     = taps * Cin;

    int ncol[4], ohv[4], owv[4], bv[4];
#pragma unroll
    for (int j = 0; j < 4; ++j) {
        ncol[j] = ntile * 64 + j * 16 + l15;
        bv[j]   = ncol[j] / HoWo;
        const int prem = ncol[j] - bv[j] * HoWo;
        ohv[j] = prem / Wo;
        owv[j] = prem - ohv[j] * Wo;
    }
    const int mrow = mtile * 16 + l15;
    const _Float16* wbase = wr + (long long)mrow * K;

    v8f acc[4] = {};
    int k0 = 0;
    for (int tp = 0; tp < taps; ++tp) {
        const int kh = tp / 3;
        const int kw2 = tp - kh * 3;
        const _Float16* xp[4];
#pragma unroll
        for (int j = 0; j < 4; ++j) {
            const int ih = ohv[j] * stride + kh - pad;
            const int iw = owv[j] * stride + kw2 - pad;
            const bool inb = (ih >= 0) & (ih < Hin) & (iw >= 0) & (iw < Win);
            xp[j] = inb ? (x + (((long long)bv[j] * Hin + ih) * Win + iw) * Cin)
                        : zpage;
        }
        for (int c = 0; c < Cin; c += 32) {
            const h8 a0 = *(const h8*)(wbase + k0 + c + half * 8);
            const h8 a1 = *(const h8*)(wbase + k0 + c + 16 + half * 8);
            const v16h af = __builtin_shufflevector(a0, a1,
                0,1,2,3,4,5,6,7,8,9,10,11,12,13,14,15);
#pragma unroll
            for (int j = 0; j < 4; ++j) {
                const h8 b0 = *(const h8*)(xp[j] + c + half * 8);
                const h8 b1 = *(const h8*)(xp[j] + c + 16 + half * 8);
                const v16h bf = __builtin_shufflevector(b0, b1,
                    0,1,2,3,4,5,6,7,8,9,10,11,12,13,14,15);
                acc[j] = __builtin_amdgcn_wmma_f32_16x16x32_f16(
                             false, af, false, bf, (short)0, acc[j], false, false);
            }
        }
        k0 += Cin;
    }

#pragma unroll
    for (int j = 0; j < 4; ++j) {
#pragma unroll
        for (int r = 0; r < 8; ++r) {
            const int m = mtile * 16 + half * 8 + r;
            float v = acc[j][r];
            if (m < Cout) v += bias[m];
            if (relu) v = fmaxf(v, 0.0f);
            if (out_f32) ((float*)out)[(long long)ncol[j] * CoutPad + m] = v;
            else ((_Float16*)out)[(long long)ncol[j] * CoutPad + m] = (_Float16)v;
        }
    }
}

// ---------------------------------------------------------------------------
// Mask GEMM: sigmoid(coef_k[112x32] x proto^T) per image; 16x64 tile per
// wave, one shared A fragment, 4 WMMAs (single K=32 step).
// ---------------------------------------------------------------------------
__global__ __launch_bounds__(32) void wmma_maskgemm_kernel(
    const _Float16* __restrict__ coefk,  // [B][112][32]
    const _Float16* __restrict__ proto,  // [B][2048][32] (NHWC from pw2)
    float* __restrict__ mout)            // [B][112][2048]
{
    const int lane = threadIdx.x;
    const int half = lane >> 4;
    const int l15  = lane & 15;
    const int mtile = blockIdx.x;   // 0..6
    const int ntile = blockIdx.y;   // 0..31 (64-wide)
    const int b     = blockIdx.z;   // 0..7
    const int mrow = mtile * 16 + l15;

    const _Float16* abase = coefk + ((long long)b * TOPKP + mrow) * PCOEF;
    const h8 a0 = *(const h8*)(abase + half * 8);
    const h8 a1 = *(const h8*)(abase + 16 + half * 8);
    const v16h af = __builtin_shufflevector(a0, a1,
        0,1,2,3,4,5,6,7,8,9,10,11,12,13,14,15);

    v8f acc[4] = {};
#pragma unroll
    for (int j = 0; j < 4; ++j) {
        const int ncol = ntile * 64 + j * 16 + l15;
        const _Float16* bbase = proto + ((long long)b * NA + ncol) * PCOEF;
        const h8 b0 = *(const h8*)(bbase + half * 8);
        const h8 b1 = *(const h8*)(bbase + 16 + half * 8);
        const v16h bf = __builtin_shufflevector(b0, b1,
            0,1,2,3,4,5,6,7,8,9,10,11,12,13,14,15);
        acc[j] = __builtin_amdgcn_wmma_f32_16x16x32_f16(
                     false, af, false, bf, (short)0, acc[j], false, false);
    }

#pragma unroll
    for (int j = 0; j < 4; ++j) {
        const int ncol = ntile * 64 + j * 16 + l15;
#pragma unroll
        for (int r = 0; r < 8; ++r) {
            const int m = mtile * 16 + half * 8 + r;
            mout[((long long)b * TOPKP + m) * NA + ncol] =
                1.0f / (1.0f + __expf(-acc[j][r]));
        }
    }
}

// ---------------------------------------------------------------------------
// Softmax over 21 classes, score/label, box decode. One thread per anchor.
// ---------------------------------------------------------------------------
__global__ void postprocess_kernel(
    const float* __restrict__ cls,   // [NTOT][32]
    const float* __restrict__ boxh,  // [NTOT][16]
    float* __restrict__ scores, int* __restrict__ labels,
    float* __restrict__ boxes)       // [NTOT][4]
{
    const int idx = blockIdx.x * blockDim.x + threadIdx.x;
    if (idx >= NTOT) return;
    const int pos = idx % NA;

    float lg[NCLS];
    float mx = -FLT_MAX;
#pragma unroll
    for (int c = 0; c < NCLS; ++c) { lg[c] = cls[(long long)idx * 32 + c]; mx = fmaxf(mx, lg[c]); }
    float sum = 0.0f;
#pragma unroll
    for (int c = 0; c < NCLS; ++c) { lg[c] = __expf(lg[c] - mx); sum += lg[c]; }
    float best = -1.0f; int bl = 0;
#pragma unroll
    for (int c = 0; c < NCLS - 1; ++c) {     // drop background (last class)
        const float p = lg[c] / sum;
        if (p > best) { best = p; bl = c; }
    }
    scores[idx] = best;
    labels[idx] = bl;

    const int ay = pos / FW, ax = pos % FW;
    const float cx = (ax + 0.5f) * 8.0f;
    const float cy = (ay + 0.5f) * 8.0f;
    const float dx = boxh[(long long)idx * 16 + 0];
    const float dy = boxh[(long long)idx * 16 + 1];
    const float dw = boxh[(long long)idx * 16 + 2];
    const float dh = boxh[(long long)idx * 16 + 3];
    const float pcx = cx + dx * 32.0f, pcy = cy + dy * 32.0f;
    const float pw  = 32.0f * __expf(dw), ph = 32.0f * __expf(dh);
    boxes[idx * 4 + 0] = pcx - pw * 0.5f;
    boxes[idx * 4 + 1] = pcy - ph * 0.5f;
    boxes[idx * 4 + 2] = pcx + pw * 0.5f;
    boxes[idx * 4 + 3] = pcy + ph * 0.5f;
}

// ---------------------------------------------------------------------------
// Greedy batched NMS (label-offset trick), one workgroup per image, LDS-resident.
// ---------------------------------------------------------------------------
__global__ __launch_bounds__(256) void nms_kernel(
    const float* __restrict__ boxes, const float* __restrict__ scores,
    const int* __restrict__ labels, int* __restrict__ keep)   // keep[B][TOPK]
{
    const int b = blockIdx.x;
    const int t = threadIdx.x;

    __shared__ float sx0[NA], sy0[NA], sx1[NA], sy1[NA], ss[NA], sarea[NA];
    __shared__ unsigned char dead[NA];
    __shared__ float red[256];
    __shared__ int   redi[256];
    __shared__ int   spick, sdone;

    float lm = -3.0e38f;
    for (int i = t; i < NA; i += 256) {
        const float* bp = boxes + ((long long)b * NA + i) * 4;
#pragma unroll
        for (int c = 0; c < 4; ++c) lm = fmaxf(lm, bp[c]);
    }
    red[t] = lm; __syncthreads();
    for (int s = 128; s > 0; s >>= 1) {
        if (t < s) red[t] = fmaxf(red[t], red[t + s]);
        __syncthreads();
    }
    const float maxb = red[0] + 1.0f;
    __syncthreads();

    for (int i = t; i < NA; i += 256) {
        const float* bp = boxes + ((long long)b * NA + i) * 4;
        const float off = (float)labels[b * NA + i] * maxb;
        const float x0 = bp[0] + off, y0 = bp[1] + off;
        const float x1 = bp[2] + off, y1 = bp[3] + off;
        sx0[i] = x0; sy0[i] = y0; sx1[i] = x1; sy1[i] = y1;
        sarea[i] = (x1 - x0) * (y1 - y0);
        ss[i] = scores[b * NA + i];
        dead[i] = 0;
    }
    if (t < TOPK) keep[b * TOPK + t] = -1;
    if (t == 0) sdone = 0;
    __syncthreads();

    for (int kept = 0; kept < TOPK; ++kept) {
        float bv = -3.0e38f; int bi = -1;
        for (int i = t; i < NA; i += 256)
            if (!dead[i] && ss[i] > bv) { bv = ss[i]; bi = i; }
        red[t] = bv; redi[t] = bi; __syncthreads();
        for (int s = 128; s > 0; s >>= 1) {
            if (t < s && red[t + s] > red[t]) { red[t] = red[t + s]; redi[t] = redi[t + s]; }
            __syncthreads();
        }
        if (t == 0) {
            if (redi[0] < 0) sdone = 1;
            else { spick = redi[0]; keep[b * TOPK + kept] = spick; dead[spick] = 1; }
        }
        __syncthreads();
        if (sdone) break;
        const int p = spick;
        const float px0 = sx0[p], py0 = sy0[p], px1 = sx1[p], py1 = sy1[p], pa = sarea[p];
        for (int i = t; i < NA; i += 256) {
            if (dead[i]) continue;
            const float ix0 = fmaxf(px0, sx0[i]), iy0 = fmaxf(py0, sy0[i]);
            const float ix1 = fminf(px1, sx1[i]), iy1 = fminf(py1, sy1[i]);
            const float iw = fmaxf(ix1 - ix0, 0.0f), ih = fmaxf(iy1 - iy0, 0.0f);
            const float inter = iw * ih;
            const float iou = inter / (pa + sarea[i] - inter + 1e-9f);
            if (iou > IOU_THR) dead[i] = 1;
        }
        __syncthreads();
    }
}

// ---------------------------------------------------------------------------
// Gather kept detections into d_out + build f16 coef_k (zero-padded to 112).
// ---------------------------------------------------------------------------
__global__ void gather_kernel(
    const float* __restrict__ boxes, const float* __restrict__ scores,
    const int* __restrict__ labels, const int* __restrict__ keep,
    const float* __restrict__ coef,           // [NTOT][32] NHWC f32
    float* __restrict__ o_boxes, float* __restrict__ o_scores,
    float* __restrict__ o_labels, float* __restrict__ o_valid,
    _Float16* __restrict__ coefk)             // [B][112][32]
{
    const int b = blockIdx.x;
    const int j = threadIdx.x;                // 0..111
    if (j >= TOPKP) return;

    if (j < TOPK) {
        const int k = keep[b * TOPK + j];
        const int valid = (k >= 0) ? 1 : 0;
        const int ki = valid ? k : 0;
        const float vf = (float)valid;
#pragma unroll
        for (int c = 0; c < 4; ++c)
            o_boxes[(b * TOPK + j) * 4 + c] = boxes[((long long)b * NA + ki) * 4 + c] * vf;
        o_scores[b * TOPK + j] = scores[b * NA + ki] * vf;
        o_labels[b * TOPK + j] = valid ? (float)labels[b * NA + ki] : 0.0f;
        o_valid[b * TOPK + j]  = vf;
        // reference gathers coef with the clipped index, unmasked
        for (int p = 0; p < PCOEF; ++p)
            coefk[((long long)b * TOPKP + j) * PCOEF + p] =
                (_Float16)coef[((long long)b * NA + ki) * PCOEF + p];
    } else {
        for (int p = 0; p < PCOEF; ++p)
            coefk[((long long)b * TOPKP + j) * PCOEF + p] = (_Float16)0.0f;
    }
}

// ---------------------------------------------------------------------------
// Bilinear upsample 32x64 -> 256x512, threshold, AND valid; NT stores for the
// ~420 MB output stream (the roofline-dominant term on 23.3 TB/s HBM).
// ---------------------------------------------------------------------------
__global__ void resize_kernel(
    const float* __restrict__ mm,     // [B][112][32*64] sigmoid values
    const int*   __restrict__ keep,   // [B][TOPK]
    float* __restrict__ o_masks)      // [B][TOPK][256][512]
{
    const long long total = (long long)BATCH * TOPK * IMG_H * IMG_W;
    long long idx = (long long)blockIdx.x * blockDim.x + threadIdx.x;
    if (idx >= total) return;

    const int ox = (int)(idx % IMG_W);
    long long tt = idx / IMG_W;
    const int oy = (int)(tt % IMG_H);
    tt /= IMG_H;
    const int j = (int)(tt % TOPK);
    const int b = (int)(tt / TOPK);

    const bool valid = keep[b * TOPK + j] >= 0;

    float sy = (oy + 0.5f) * 0.125f - 0.5f;
    float sx = (ox + 0.5f) * 0.125f - 0.5f;
    sy = fminf(fmaxf(sy, 0.0f), (float)(FH - 1));
    sx = fminf(fmaxf(sx, 0.0f), (float)(FW - 1));
    const int y0 = (int)floorf(sy), x0 = (int)floorf(sx);
    const int y1 = min(y0 + 1, FH - 1), x1 = min(x0 + 1, FW - 1);
    const float wy = sy - (float)y0, wx = sx - (float)x0;

    const float* base = mm + ((long long)b * TOPKP + j) * NA;
    const float v00 = base[y0 * FW + x0], v01 = base[y0 * FW + x1];
    const float v10 = base[y1 * FW + x0], v11 = base[y1 * FW + x1];
    const float val = (v00 * (1.0f - wy) + v10 * wy) * (1.0f - wx)
                    + (v01 * (1.0f - wy) + v11 * wy) * wx;
    const float outv = (val > 0.5f && valid) ? 1.0f : 0.0f;
    __builtin_nontemporal_store(outv, &o_masks[idx]);
}

// ---------------------------------------------------------------------------
extern "C" void kernel_launch(void* const* d_in, const int* in_sizes, int n_in,
                              void* d_out, int out_size, void* d_ws, size_t ws_size,
                              hipStream_t stream) {
    (void)in_sizes; (void)n_in; (void)out_size; (void)ws_size;
    const float* images = (const float*)d_in[0];
    const float* w1  = (const float*)d_in[1];  const float* b1  = (const float*)d_in[2];
    const float* w2  = (const float*)d_in[3];  const float* b2  = (const float*)d_in[4];
    const float* w3  = (const float*)d_in[5];  const float* b3  = (const float*)d_in[6];
    const float* pw1 = (const float*)d_in[7];  const float* pb1 = (const float*)d_in[8];
    const float* pw2 = (const float*)d_in[9];  const float* pb2 = (const float*)d_in[10];
    const float* cw  = (const float*)d_in[11]; const float* cb  = (const float*)d_in[12];
    const float* bw  = (const float*)d_in[13]; const float* bb  = (const float*)d_in[14];
    const float* kw  = (const float*)d_in[15]; const float* kb  = (const float*)d_in[16];

    const long long N1 = 8LL * 128 * 256;   // 262144
    const long long N2 = 8LL * 64 * 128;    // 65536
    const long long N3 = 8LL * 32 * 64;     // 16384 == NTOT

    // ---- workspace layout ----
    char* wsb = (char*)d_ws;
    long long o = 0;   // bytes
    auto alloc = [&](long long bytes) { char* p = wsb + o; o += (bytes + 255) & ~255LL; return (void*)p; };
    _Float16* X1    = (_Float16*)alloc(N1 * 64  * 2);   // [B][128][256][64]
    _Float16* X2    = (_Float16*)alloc(N2 * 128 * 2);   // [B][64][128][128]
    _Float16* FEAT  = (_Float16*)alloc(N3 * 256 * 2);   // [B][32][64][256]
    _Float16* PH    = (_Float16*)alloc(N3 * 256 * 2);
    _Float16* PROTO = (_Float16*)alloc(N3 * 32  * 2);   // [B][2048][32]
    float*    CLS   = (float*)alloc(N3 * 32 * 4);       // [NTOT][32]
    float*    BOXH  = (float*)alloc(N3 * 16 * 4);       // [NTOT][16]
    float*    COEF  = (float*)alloc(N3 * 32 * 4);       // [NTOT][32]
    _Float16* WR2   = (_Float16*)alloc(128LL * 576 * 2);
    _Float16* WR3   = (_Float16*)alloc(256LL * 1152 * 2);
    _Float16* WRP1  = (_Float16*)alloc(256LL * 2304 * 2);
    _Float16* WRP2  = (_Float16*)alloc(32LL * 256 * 2);
    _Float16* WRC   = (_Float16*)alloc(32LL * 2304 * 2);
    _Float16* WRB   = (_Float16*)alloc(16LL * 2304 * 2);
    _Float16* WRK   = (_Float16*)alloc(32LL * 2304 * 2);
    _Float16* ZPAGE = (_Float16*)alloc(ZPAGE_H * 2);
    float*    SCOR  = (float*)alloc(N3 * 4);
    int*      LABL  = (int*)alloc(N3 * 4);
    float*    BOXS  = (float*)alloc(N3 * 4 * 4);
    int*      KEEP  = (int*)alloc(BATCH * TOPK * 4);
    _Float16* COEFK = (_Float16*)alloc((long long)BATCH * TOPKP * PCOEF * 2);
    float*    MM    = (float*)alloc((long long)BATCH * TOPKP * NA * 4);

    // ---- d_out layout (reference tuple, flattened in return order) ----
    float* out      = (float*)d_out;
    float* o_boxes  = out;                                   // 8*100*4
    float* o_scores = out + 3200;                            // 8*100
    float* o_labels = out + 4000;                            // 8*100
    float* o_masks  = out + 4800;                            // 8*100*256*512
    float* o_valid  = out + 4800 + (long long)BATCH * TOPK * IMG_H * IMG_W;

    // ---- init zero page + weight repack ----
    zero_init_kernel<<<(ZPAGE_H + 255) / 256, 256, 0, stream>>>(ZPAGE, ZPAGE_H);
    auto rp = [&](const float* w, _Float16* wr, int Cout, int CoutPad, int Cin, int taps) {
        const long long tot = (long long)CoutPad * taps * Cin;
        repack_w_kernel<<<(unsigned)((tot + 255) / 256), 256, 0, stream>>>(
            w, wr, Cout, CoutPad, Cin, taps);
    };
    rp(w2,  WR2, 128, 128, 64, 9);
    rp(w3,  WR3, 256, 256, 128, 9);
    rp(pw1, WRP1, 256, 256, 256, 9);
    rp(pw2, WRP2, 32, 32, 256, 1);
    rp(cw,  WRC, 21, 32, 256, 9);
    rp(bw,  WRB, 4, 16, 256, 9);
    rp(kw,  WRK, 32, 32, 256, 9);

    // ---- backbone ----
    conv1_direct_kernel<<<(unsigned)((N1 * 64 + 255) / 256), 256, 0, stream>>>(
        images, w1, b1, X1);
    // conv2: 64->128, s2, 128x256 -> 64x128
    wmma_conv_nhwc_kernel<<<dim3(8, (unsigned)(N2 / 64)), 32, 0, stream>>>(
        X1, WR2, b2, ZPAGE, X2, 0, 64, 128, 128, 128, 256, 64, 128, 2, 1, 9, 1);
    // conv3: 128->256, s2, 64x128 -> 32x64
    wmma_conv_nhwc_kernel<<<dim3(16, (unsigned)(N3 / 64)), 32, 0, stream>>>(
        X2, WR3, b3, ZPAGE, FEAT, 0, 128, 256, 256, 64, 128, 32, 64, 2, 1, 9, 1);
    // proto: pw1 3x3 relu -> PH ; pw2 1x1 -> PROTO (f16 NHWC)
    wmma_conv_nhwc_kernel<<<dim3(16, (unsigned)(N3 / 64)), 32, 0, stream>>>(
        FEAT, WRP1, pb1, ZPAGE, PH, 0, 256, 256, 256, 32, 64, 32, 64, 1, 1, 9, 1);
    wmma_conv_nhwc_kernel<<<dim3(2, (unsigned)(N3 / 64)), 32, 0, stream>>>(
        PH, WRP2, pb2, ZPAGE, PROTO, 0, 256, 32, 32, 32, 64, 32, 64, 1, 0, 1, 0);
    // heads (f32 NHWC outputs)
    wmma_conv_nhwc_kernel<<<dim3(2, (unsigned)(N3 / 64)), 32, 0, stream>>>(
        FEAT, WRC, cb, ZPAGE, CLS, 1, 256, 21, 32, 32, 64, 32, 64, 1, 1, 9, 0);
    wmma_conv_nhwc_kernel<<<dim3(1, (unsigned)(N3 / 64)), 32, 0, stream>>>(
        FEAT, WRB, bb, ZPAGE, BOXH, 1, 256, 4, 16, 32, 64, 32, 64, 1, 1, 9, 0);
    wmma_conv_nhwc_kernel<<<dim3(2, (unsigned)(N3 / 64)), 32, 0, stream>>>(
        FEAT, WRK, kb, ZPAGE, COEF, 1, 256, 32, 32, 32, 64, 32, 64, 1, 1, 9, 0);

    // ---- detection post-processing ----
    postprocess_kernel<<<NTOT / 256, 256, 0, stream>>>(CLS, BOXH, SCOR, LABL, BOXS);
    nms_kernel<<<BATCH, 256, 0, stream>>>(BOXS, SCOR, LABL, KEEP);
    gather_kernel<<<BATCH, TOPKP, 0, stream>>>(
        BOXS, SCOR, LABL, KEEP, COEF, o_boxes, o_scores, o_labels, o_valid, COEFK);

    // ---- masks ----
    wmma_maskgemm_kernel<<<dim3(TOPKP / 16, NA / 64, BATCH), 32, 0, stream>>>(
        COEFK, PROTO, MM);
    const long long total_mask = (long long)BATCH * TOPK * IMG_H * IMG_W;
    resize_kernel<<<(unsigned)((total_mask + 255) / 256), 256, 0, stream>>>(
        MM, KEEP, o_masks);
}
